// UlyssesAttentionProcessor_68006512165412
// MI455X (gfx1250) — matrix operand order repaired and checked by reading, not compiled, not run on hardware
//
#include <hip/hip_runtime.h>
#include <hip/hip_bf16.h>
#include <stdint.h>

#define HID   2048
#define MROWS 4096   // B*S = 2*2048
#define HEADS 16
#define HDIM  128
#define BH    32     // B*HEADS

typedef __attribute__((ext_vector_type(16))) __bf16        v16bf;
typedef __attribute__((ext_vector_type(8)))  float         v8f;
typedef __attribute__((ext_vector_type(4)))  unsigned int  u32x4;
typedef __attribute__((ext_vector_type(4)))  float         f32x4;
typedef __attribute__((ext_vector_type(4)))  unsigned short u16x4;

union Frag { v16bf v; u32x4 q[2]; };

__device__ __forceinline__ unsigned short f32_to_bf16(float f) {
  unsigned int u = __builtin_bit_cast(unsigned int, f);
  u += 0x7FFFu + ((u >> 16) & 1u);      // round-to-nearest-even
  return (unsigned short)(u >> 16);
}

// A fragment (16x32 bf16, MxK): lanes 0-15 row m=lane, K [k0..k0+15]
//                               lanes 16-31 row m=lane-16, K [k0+16..k0+31]
__device__ __forceinline__ v16bf load_a_frag(const unsigned short* base,
                                             int ld, int mr, int k0, int lane) {
  int row = mr + (lane & 15);
  int kc  = k0 + ((lane >> 4) << 4);
  const u32x4* p = (const u32x4*)(base + (size_t)row * ld + kc);
  Frag f; f.q[0] = p[0]; f.q[1] = p[1];
  return f.v;
}

// B fragment (32x16 bf16, KxN) with B[k][n] = src[n][k] (src row-major [N,K]):
// lanes 0-15: n=n0+lane, K [k0..k0+15]; lanes 16-31: n=n0+lane-16, K [k0+16..k0+31]
__device__ __forceinline__ v16bf load_b_frag(const unsigned short* base,
                                             int ld, int n0, int k0, int lane) {
  int row = n0 + (lane & 15);
  int kc  = k0 + ((lane >> 4) << 4);
  const u32x4* p = (const u32x4*)(base + (size_t)row * ld + kc);
  Frag f; f.q[0] = p[0]; f.q[1] = p[1];
  return f.v;
}

__device__ __forceinline__ v8f wmma_bf16(v16bf a, v16bf b, v8f c) {
  return __builtin_amdgcn_wmma_f32_16x16x32_bf16(false, a, false, b, (short)0, c,
                                                 false, false);
}

// async global -> LDS, 16 bytes, ASYNCcnt-tracked (CDNA5 path)
__device__ __forceinline__ void async_load_16B(unsigned lds_off, const unsigned short* g) {
  asm volatile("global_load_async_to_lds_b128 %0, %1, off"
               :: "v"(lds_off), "v"((unsigned long long)(uintptr_t)g)
               : "memory");
}

// ---------------------------------------------------------------- convert f32 -> bf16
__global__ __launch_bounds__(256)
void cvt_f32_bf16_kernel(const float* __restrict__ in, unsigned short* __restrict__ out,
                         int n4) {
  int i = blockIdx.x * 256 + threadIdx.x;
  if (i >= n4) return;
  f32x4 f = ((const f32x4*)in)[i];
  u16x4 r;
#pragma unroll
  for (int j = 0; j < 4; ++j) r[j] = f32_to_bf16(f[j]);
  ((u16x4*)out)[i] = r;
}

// ---------------------------------------------------------------- GEMM  Y = (A(bf16) @ W^T + bias) * outScale
// A: [MROWS, HID] bf16 row-major; W: [HID, HID] bf16 row-major ([N,K])
// Block tile 128x64, K-step 64, double-buffered async global->LDS staging.
// MODE 0: f32 out [MROWS, HID]
// MODE 1: bf16 out, head layout [BH][S][HDIM]            (Q, K)
// MODE 2: bf16 out, transposed head layout [BH][HDIM][S] (V)
template <int MODE>
__global__ __launch_bounds__(256)
void gemm_bias_kernel(const unsigned short* __restrict__ A,
                      const unsigned short* __restrict__ W,
                      const float* __restrict__ bias,
                      float outScale,
                      void* __restrict__ out) {
  __shared__ __align__(128) unsigned short tA[2][128 * 64];  // 2 x 16 KB
  __shared__ __align__(128) unsigned short tB[2][64 * 64];   // 2 x  8 KB

  const int tid  = threadIdx.x;
  const int lane = tid & 31;
  const int wave = tid >> 5;
  const int wm = wave & 3;           // 4 waves along M
  const int wn = wave >> 2;          // 2 waves along N
  const int mB = blockIdx.y * 128;
  const int nB = blockIdx.x * 64;
  const int m0 = mB + wm * 32;
  const int n0 = nB + wn * 32;

  const unsigned aBase[2] = { (unsigned)(uintptr_t)&tA[0][0],
                              (unsigned)(uintptr_t)&tA[1][0] };
  const unsigned bBase[2] = { (unsigned)(uintptr_t)&tB[0][0],
                              (unsigned)(uintptr_t)&tB[1][0] };

  // stage one K-step (64 columns) of A (128x64) and B (64x64) into LDS buffer `buf`
  auto issue_tiles = [&](int buf, int k0) {
#pragma unroll
    for (int j = 0; j < 4; ++j) {                 // A: 1024 16B chunks / 256 thr
      int c = tid + j * 256;
      int r = c >> 3, q = c & 7;
      async_load_16B(aBase[buf] + (unsigned)(r * 64 + q * 8) * 2,
                     A + (size_t)(mB + r) * HID + k0 + q * 8);
    }
#pragma unroll
    for (int j = 0; j < 2; ++j) {                 // B: 512 16B chunks / 256 thr
      int c = tid + j * 256;
      int r = c >> 3, q = c & 7;
      async_load_16B(bBase[buf] + (unsigned)(r * 64 + q * 8) * 2,
                     W + (size_t)(nB + r) * HID + k0 + q * 8);
    }
  };

  v8f acc00 = {}, acc01 = {}, acc10 = {}, acc11 = {};

  issue_tiles(0, 0);
#pragma unroll 1
  for (int k0 = 0; k0 < HID; k0 += 64) {
    const int cur = (k0 >> 6) & 1;
    asm volatile("s_wait_asynccnt 0" ::: "memory");  // own staging loads complete
    __syncthreads();                                  // whole tile visible; prev buffer free
    if (k0 + 64 < HID) issue_tiles(1 - cur, k0 + 64); // overlap next stage with WMMA
    const unsigned short* tAc = &tA[cur][0];
    const unsigned short* tBc = &tB[cur][0];
#pragma unroll
    for (int kc = 0; kc < 64; kc += 32) {
      v16bf a0 = load_a_frag(tAc, 64, wm * 32,      kc, lane);
      v16bf a1 = load_a_frag(tAc, 64, wm * 32 + 16, kc, lane);
      v16bf b0 = load_b_frag(tBc, 64, wn * 32,      kc, lane);
      v16bf b1 = load_b_frag(tBc, 64, wn * 32 + 16, kc, lane);
      acc00 = wmma_bf16(a0, b0, acc00);
      acc01 = wmma_bf16(a0, b1, acc01);
      acc10 = wmma_bf16(a1, b0, acc10);
      acc11 = wmma_bf16(a1, b1, acc11);
    }
  }

  // epilogue: C/D layout — VGPR v: row = v + 8*(lane>=16), col = lane&15 (per 16x16 tile)
  const int rsub = ((lane >> 4) << 3);
  const int csub = lane & 15;
#pragma unroll
  for (int mi = 0; mi < 2; ++mi) {
#pragma unroll
    for (int nj = 0; nj < 2; ++nj) {
      v8f acc = (mi == 0) ? (nj == 0 ? acc00 : acc01) : (nj == 0 ? acc10 : acc11);
      int colg = n0 + nj * 16 + csub;
      float bv = bias[colg];
#pragma unroll
      for (int v = 0; v < 8; ++v) {
        int rowg = m0 + mi * 16 + v + rsub;
        float val = (acc[v] + bv) * outScale;
        if (MODE == 0) {
          ((float*)out)[(size_t)rowg * HID + colg] = val;
        } else {
          int b = rowg >> 11, s = rowg & 2047;
          int h = colg >> 7,  d = colg & 127;
          unsigned short bf = f32_to_bf16(val);
          if (MODE == 1)
            ((unsigned short*)out)[(((size_t)(b * HEADS + h)) * 2048 + s) * HDIM + d] = bf;
          else
            ((unsigned short*)out)[(((size_t)(b * HEADS + h)) * HDIM + d) * 2048 + s] = bf;
        }
      }
    }
  }
}

// ---------------------------------------------------------------- flash attention
// Q (pre-scaled by 1/sqrt(D)), K: [BH][2048][128] bf16 ; Vt: [BH][HDIM][2048] bf16
// Ctx: [4096][2048] bf16
// One block = 8 waves, all on the SAME head (b,h); K/V tiles staged once per
// block per 32-key step via async global->LDS (double-buffered), shared by all
// 8 waves (8x global traffic reduction vs per-wave fetch).
__global__ __launch_bounds__(256)
void attn_kernel(const unsigned short* __restrict__ Q,
                 const unsigned short* __restrict__ K,
                 const unsigned short* __restrict__ Vt,
                 unsigned short* __restrict__ Ctx) {
  __shared__ __align__(128) unsigned short kTile[2][32 * 128]; // keys x d   (2 x 8 KB)
  __shared__ __align__(128) unsigned short vTile[2][128 * 32]; // d    x key (2 x 8 KB)
  __shared__ __align__(64)  unsigned short pshare[8][16 * 32]; // per-wave P (8 KB)

  const int tid  = threadIdx.x;
  const int lane = tid & 31;
  const int wave = tid >> 5;
  const int bh   = blockIdx.x >> 4;                     // 16 blocks per head
  const int q0   = ((blockIdx.x & 15) * 8 + wave) * 16; // query tile start

  const unsigned short* Qp = Q  + ((size_t)bh * 2048 + q0) * HDIM;
  const unsigned short* Kp = K  + (size_t)bh * 2048 * HDIM;
  const unsigned short* Vp = Vt + (size_t)bh * HDIM * 2048;

  const unsigned kBase[2] = { (unsigned)(uintptr_t)&kTile[0][0],
                              (unsigned)(uintptr_t)&kTile[1][0] };
  const unsigned vBase[2] = { (unsigned)(uintptr_t)&vTile[0][0],
                              (unsigned)(uintptr_t)&vTile[1][0] };

  // stage K tile (32 keys x 128 d) and V tile (128 d x 32 keys) for key block kk
  auto issue_kv = [&](int buf, int kk) {
#pragma unroll
    for (int j = 0; j < 2; ++j) {               // K: 512 16B chunks / 256 thr
      int c = tid + j * 256;
      int r = c >> 4, q = c & 15;               // r = key 0..31, q = d-chunk
      async_load_16B(kBase[buf] + (unsigned)(r * 128 + q * 8) * 2,
                     Kp + (size_t)(kk + r) * HDIM + q * 8);
    }
#pragma unroll
    for (int j = 0; j < 2; ++j) {               // V: 512 16B chunks / 256 thr
      int c = tid + j * 256;
      int r = c >> 2, q = c & 3;                // r = d 0..127, q = key-chunk
      async_load_16B(vBase[buf] + (unsigned)(r * 32 + q * 8) * 2,
                     Vp + (size_t)r * 2048 + kk + q * 8);
    }
  };

  v16bf aq[4];
#pragma unroll
  for (int dc = 0; dc < 4; ++dc) aq[dc] = load_a_frag(Qp, HDIM, 0, dc * 32, lane);

  v8f o[8] = {};
  float rmax[8], rsum[8];
#pragma unroll
  for (int i = 0; i < 8; ++i) { rmax[i] = -3.0e38f; rsum[i] = 0.0f; }

  const int rsub = ((lane >> 4) << 3);
  const int csub = lane & 15;
  unsigned short* pbase = &pshare[wave][0];

  issue_kv(0, 0);
#pragma unroll 1
  for (int kk = 0; kk < 2048; kk += 32) {
    const int cur = (kk >> 5) & 1;
    asm volatile("s_wait_asynccnt 0" ::: "memory");   // own staging loads complete
    __syncthreads();                                   // tiles visible; prev buffer free
    if (kk + 32 < 2048) issue_kv(1 - cur, kk + 32);    // overlap next stage
    const unsigned short* kT = &kTile[cur][0];
    const unsigned short* vT = &vTile[cur][0];

    // scores: 16 queries x 32 keys (Q already carries the 1/sqrt(D) scale)
    v8f s0 = {}, s1 = {};
#pragma unroll
    for (int dc = 0; dc < 4; ++dc) {
      s0 = wmma_bf16(aq[dc], load_b_frag(kT, HDIM, 0,  dc * 32, lane), s0);
      s1 = wmma_bf16(aq[dc], load_b_frag(kT, HDIM, 16, dc * 32, lane), s1);
    }
    // online softmax per row (row i lives in VGPR i across a 16-lane group)
#pragma unroll
    for (int i = 0; i < 8; ++i) {
      float a = s0[i];
      float b = s1[i];
      float t = fmaxf(a, b);
      t = fmaxf(t, __shfl_xor(t, 1, 32));
      t = fmaxf(t, __shfl_xor(t, 2, 32));
      t = fmaxf(t, __shfl_xor(t, 4, 32));
      t = fmaxf(t, __shfl_xor(t, 8, 32));
      float nm    = fmaxf(rmax[i], t);
      float alpha = __expf(rmax[i] - nm);
      rmax[i] = nm;
      float p0 = __expf(a - nm);
      float p1 = __expf(b - nm);
      float ps = p0 + p1;
      ps += __shfl_xor(ps, 1, 32);
      ps += __shfl_xor(ps, 2, 32);
      ps += __shfl_xor(ps, 4, 32);
      ps += __shfl_xor(ps, 8, 32);
      rsum[i] = rsum[i] * alpha + ps;
#pragma unroll
      for (int dt = 0; dt < 8; ++dt) o[dt][i] *= alpha;
      // stage P (C layout -> LDS, row-major 16x32 bf16, per-wave region)
      int r = i + rsub;
      pbase[r * 32 + csub]      = f32_to_bf16(p0);
      pbase[r * 32 + csub + 16] = f32_to_bf16(p1);
    }
    asm volatile("s_wait_dscnt 0" ::: "memory");
    // reload P in A-matrix layout (ds_load_b128 x2)
    const u32x4* pp = (const u32x4*)(pbase + (size_t)(lane & 15) * 32 + ((lane >> 4) << 4));
    Frag fp; fp.q[0] = pp[0]; fp.q[1] = pp[1];
    v16bf ap = fp.v;
    // O += P(16x32) @ V(32x128) from staged V tile
#pragma unroll
    for (int dt = 0; dt < 8; ++dt)
      o[dt] = wmma_bf16(ap, load_b_frag(vT, 32, dt * 16, 0, lane), o[dt]);
  }

  // normalize and write context, layout [B][S][H*D]
  const int b = bh >> 4, h = bh & 15;
#pragma unroll
  for (int i = 0; i < 8; ++i) {
    float inv = 1.0f / rsum[i];
    int s = q0 + i + rsub;
#pragma unroll
    for (int dt = 0; dt < 8; ++dt) {
      int col = h * HDIM + dt * 16 + csub;
      Ctx[((size_t)b * 2048 + s) * HID + col] = f32_to_bf16(o[dt][i] * inv);
    }
  }
}

// ---------------------------------------------------------------- launch
extern "C" void kernel_launch(void* const* d_in, const int* in_sizes, int n_in,
                              void* d_out, int out_size, void* d_ws, size_t ws_size,
                              hipStream_t stream) {
  const float* X  = (const float*)d_in[0];
  const float* wq = (const float*)d_in[1];
  const float* bq = (const float*)d_in[2];
  const float* wk = (const float*)d_in[3];
  const float* bk = (const float*)d_in[4];
  const float* wv = (const float*)d_in[5];
  const float* bv = (const float*)d_in[6];
  const float* wo = (const float*)d_in[7];
  const float* bo = (const float*)d_in[8];

  char* ws = (char*)d_ws;
  const size_t SZ_X = (size_t)MROWS * HID * 2;   // 16 MB bf16
  const size_t SZ_W = (size_t)HID * HID * 2;     //  8 MB bf16
  unsigned short* Xb  = (unsigned short*)(ws);
  unsigned short* Wqb = (unsigned short*)(ws + SZ_X);
  unsigned short* Wkb = (unsigned short*)(ws + SZ_X + SZ_W);
  unsigned short* Wvb = (unsigned short*)(ws + SZ_X + 2 * SZ_W);
  unsigned short* Wob = (unsigned short*)(ws + SZ_X + 3 * SZ_W);
  unsigned short* Qb  = (unsigned short*)(ws + SZ_X + 4 * SZ_W);
  unsigned short* Kb  = (unsigned short*)(ws + 2 * SZ_X + 4 * SZ_W);
  unsigned short* Vtb = (unsigned short*)(ws + 3 * SZ_X + 4 * SZ_W);
  unsigned short* Cb  = (unsigned short*)(ws + 4 * SZ_X + 4 * SZ_W);

  const int nX4 = MROWS * HID / 4;   // 2,097,152
  const int nW4 = HID * HID / 4;     // 1,048,576
  cvt_f32_bf16_kernel<<<nX4 / 256, 256, 0, stream>>>(X,  Xb,  nX4);
  cvt_f32_bf16_kernel<<<nW4 / 256, 256, 0, stream>>>(wq, Wqb, nW4);
  cvt_f32_bf16_kernel<<<nW4 / 256, 256, 0, stream>>>(wk, Wkb, nW4);
  cvt_f32_bf16_kernel<<<nW4 / 256, 256, 0, stream>>>(wv, Wvb, nW4);
  cvt_f32_bf16_kernel<<<nW4 / 256, 256, 0, stream>>>(wo, Wob, nW4);

  const float qscale = 0.08838834764831845f;  // 1/sqrt(128), folded into Q
  dim3 gg(HID / 64, MROWS / 128);             // (32, 32)
  gemm_bias_kernel<1><<<gg, 256, 0, stream>>>(Xb, Wqb, bq, qscale, (void*)Qb);
  gemm_bias_kernel<1><<<gg, 256, 0, stream>>>(Xb, Wkb, bk, 1.0f,   (void*)Kb);
  gemm_bias_kernel<2><<<gg, 256, 0, stream>>>(Xb, Wvb, bv, 1.0f,   (void*)Vtb);

  attn_kernel<<<(BH * 128) / 8, 256, 0, stream>>>(Qb, Kb, Vtb, Cb);

  gemm_bias_kernel<0><<<gg, 256, 0, stream>>>(Cb, Wob, bo, 1.0f, d_out);
}
